// SingleHeadSelfAttention_79766132622088
// MI455X (gfx1250) — compile-verified, matrix operand order checked by reading, not run on hardware
//
#include <hip/hip_runtime.h>
#include <hip/hip_bf16.h>

typedef __attribute__((ext_vector_type(16))) __bf16 v16bf;
typedef __attribute__((ext_vector_type(8)))  __bf16 v8bf;
typedef __attribute__((ext_vector_type(8)))  float  v8f;
typedef __attribute__((ext_vector_type(4)))  float  v4f;

#define BM 128
#define BN 128
#define BK 32
#define LDT 40                     // padded LDS row stride (bf16): 80B, 16B-aligned
#define A_ELEMS (BM * LDT)         // 5120 bf16
#define B_ELEMS (BN * LDT)         // 5120 bf16
#define STAGE   (A_ELEMS + B_ELEMS)

// Low 32 bits of a flat shared pointer are the LDS byte address
// (the LDS aperture base occupies addr[63:32] only).
static __device__ __forceinline__ unsigned lds_addr32(const void* p) {
  return (unsigned)(uintptr_t)p;
}

// CDNA5 async copy: global -> LDS, 16B per lane, tracked by ASYNCcnt.
// GVS addressing: mem = SGPR_base + VGPR_off(i32) + ioffset.
static __device__ __forceinline__ void async_b128(__bf16* lds_dst,
                                                  const __bf16* gbase,
                                                  unsigned byte_off) {
  asm volatile("global_load_async_to_lds_b128 %0, %1, %2 offset:0"
               :: "v"(lds_addr32(lds_dst)), "v"(byte_off), "s"(gbase)
               : "memory");
}

// 16x(K=32) bf16 WMMA fragment: lane holds K = b8+0..7 and b8+16..23,
// b8 = 8*(lane>>4)  ->  two contiguous 16B runs (two ds_load_b128).
static __device__ __forceinline__ v16bf frag_from_lds(const __bf16* rowbase, int b8) {
  v8bf lo = *(const v8bf*)(rowbase + b8);
  v8bf hi = *(const v8bf*)(rowbase + b8 + 16);
  v16bf r;
#pragma unroll
  for (int e = 0; e < 8; ++e) { r[e] = lo[e]; r[e + 8] = hi[e]; }
  return r;
}

#define WMMA_BF16(A, B, C) \
  __builtin_amdgcn_wmma_f32_16x16x32_bf16(false, (A), false, (B), (short)0, (C), false, false)

// Issue one pipeline stage: A tile 128x32 + B tile 128x32 bf16.
// 4 async instructions per thread (A:2, B:2) -> asynccnt += 4 per wave.
static __device__ __forceinline__ void issue_stage(const __bf16* Ag, int lda,
                                                   const __bf16* Bg, int ldb,
                                                   int k0, __bf16* Asm, __bf16* Bsm,
                                                   int tid) {
#pragma unroll
  for (int it = 0; it < 2; ++it) {
    int idx = tid + it * 256;
    int row = idx >> 2, c8 = idx & 3;
    async_b128(Asm + row * LDT + c8 * 8, Ag, (unsigned)((row * lda + k0 + c8 * 8) * 2));
  }
#pragma unroll
  for (int it = 0; it < 2; ++it) {
    int idx = tid + it * 256;
    int n = idx >> 2, c8 = idx & 3;
    async_b128(Bsm + n * LDT + c8 * 8, Bg, (unsigned)((n * ldb + k0 + c8 * 8) * 2));
  }
}

// 128x128 block, 8 waves (2 M x 4 N), wave tile 64x32 -> 8 WMMA / K-step.
// 3-stage async LDS pipeline, one barrier per K-step.
static __device__ __forceinline__ void gemm_core(const __bf16* Ag, int lda,
                                                 const __bf16* Bg, int ldb,
                                                 int kdim, __bf16* smem,
                                                 int tid, v8f c[4][2]) {
  const int lane = tid & 31, wave = tid >> 5;
  const int wm = wave >> 2, wn = wave & 3;
  const int r = lane & 15, b8 = (lane >> 4) << 3;
  const int nk = kdim / BK;   // >= 32 for all call sites

  issue_stage(Ag, lda, Bg, ldb, 0, smem, smem + A_ELEMS, tid);
  issue_stage(Ag, lda, Bg, ldb, BK, smem + STAGE, smem + STAGE + A_ELEMS, tid);

  for (int i = 0; i < nk; ++i) {
    // Retire stage i (async "done" returns in order): keep stage i+1 in flight.
    if (i + 1 < nk) asm volatile("s_wait_asynccnt 4" ::: "memory");
    else            asm volatile("s_wait_asynccnt 0" ::: "memory");
    __syncthreads();  // stage i visible to all; stage i-1 reads finished by all
    if (i + 2 < nk) {
      __bf16* st = smem + ((i + 2) % 3) * STAGE;   // == buffer read at stage i-1
      issue_stage(Ag, lda, Bg, ldb, (i + 2) * BK, st, st + A_ELEMS, tid);
    }
    const __bf16* As = smem + (i % 3) * STAGE;
    const __bf16* Bs = As + A_ELEMS;
    v16bf a[4], bb[2];
#pragma unroll
    for (int x = 0; x < 4; ++x) a[x]  = frag_from_lds(As + (wm * 64 + x * 16 + r) * LDT, b8);
#pragma unroll
    for (int y = 0; y < 2; ++y) bb[y] = frag_from_lds(Bs + (wn * 32 + y * 16 + r) * LDT, b8);
#pragma unroll
    for (int x = 0; x < 4; ++x)
#pragma unroll
      for (int y = 0; y < 2; ++y)
        c[x][y] = WMMA_BF16(a[x], bb[y], c[x][y]);
  }
}

// ---------------------------------------------------------------------------
// Prep 1: x fp32 -> bf16 (same layout).  8M elements, 8/thread.
// ---------------------------------------------------------------------------
__global__ __launch_bounds__(256) void cvt_x_kernel(const float* __restrict__ x,
                                                    __bf16* __restrict__ xb) {
  size_t base = ((size_t)blockIdx.x * 256 + threadIdx.x) * 8;
  v4f a = *(const v4f*)(x + base);
  v4f b = *(const v4f*)(x + base + 4);
  v8bf o;
#pragma unroll
  for (int i = 0; i < 4; ++i) { o[i] = (__bf16)a[i]; o[i + 4] = (__bf16)b[i]; }
  *(v8bf*)(xb + base) = o;
}

// ---------------------------------------------------------------------------
// Prep 2: Wt[n][k] = bf16(W[k][n]) for the three 1024x1024 weights. grid=(16,16,3)
// ---------------------------------------------------------------------------
__global__ __launch_bounds__(256) void cvt_w_kernel(
    const float* __restrict__ W0, const float* __restrict__ W1, const float* __restrict__ W2,
    __bf16* __restrict__ T0, __bf16* __restrict__ T1, __bf16* __restrict__ T2) {
  __shared__ __bf16 tile[64 * 72];   // [n][k], padded
  const int tid = threadIdx.x;
  const int k0 = blockIdx.x * 64, n0 = blockIdx.y * 64;
  const float* W = (blockIdx.z == 0) ? W0 : (blockIdx.z == 1) ? W1 : W2;
  __bf16* T      = (blockIdx.z == 0) ? T0 : (blockIdx.z == 1) ? T1 : T2;
#pragma unroll
  for (int it = 0; it < 4; ++it) {
    int idx = tid + it * 256;
    int kk = idx >> 4, c4 = idx & 15;
    v4f w = *(const v4f*)(W + (size_t)(k0 + kk) * 1024 + n0 + c4 * 4);
#pragma unroll
    for (int i = 0; i < 4; ++i) tile[(c4 * 4 + i) * 72 + kk] = (__bf16)w[i];
  }
  __syncthreads();
#pragma unroll
  for (int it = 0; it < 2; ++it) {
    int idx = tid + it * 256;
    int n = idx >> 3, c8 = idx & 7;
    *(v8bf*)(T + (size_t)(n0 + n) * 1024 + k0 + c8 * 8) = *(const v8bf*)(tile + n * 72 + c8 * 8);
  }
}

// ---------------------------------------------------------------------------
// Kernel 1: QKV projection.  grid=(64,8,3).  V stored transposed: Vt[b][d][l].
// ---------------------------------------------------------------------------
__global__ __launch_bounds__(256) void proj_kernel(
    const __bf16* __restrict__ xb,
    const __bf16* __restrict__ Wtq, const __bf16* __restrict__ Wtk, const __bf16* __restrict__ Wtv,
    const float* __restrict__ bq, const float* __restrict__ bk, const float* __restrict__ bv,
    __bf16* __restrict__ Q, __bf16* __restrict__ K, __bf16* __restrict__ Vt) {
  __shared__ __align__(16) __bf16 smem[3 * STAGE];
  const int tid = threadIdx.x;
  const int rowBase = blockIdx.x * BM, colBase = blockIdx.y * BN;
  const int which = blockIdx.z;
  const __bf16* Wt  = (which == 0) ? Wtq : (which == 1) ? Wtk : Wtv;
  const float* bias = (which == 0) ? bq : (which == 1) ? bk : bv;
  const float scale = (which == 0) ? 0.03125f : 1.0f;   // 1/sqrt(1024) folded into Q

  v8f c[4][2] = {};
  gemm_core(xb + (size_t)rowBase * 1024, 1024,
            Wt + (size_t)colBase * 1024, 1024, 1024, smem, tid, c);

  const int lane = tid & 31, wave = tid >> 5;
  const int wm = wave >> 2, wn = wave & 3;
  const int row0 = rowBase + wm * 64 + ((lane >> 4) << 3);
  const int col0 = colBase + wn * 32 + (lane & 15);
#pragma unroll
  for (int x = 0; x < 4; ++x)
#pragma unroll
    for (int y = 0; y < 2; ++y) {
      int col = col0 + y * 16;
      float bv_ = bias[col];
#pragma unroll
      for (int v = 0; v < 8; ++v) {
        int row = row0 + x * 16 + v;
        float val = (c[x][y][v] + bv_) * scale;
        if (which == 0)      Q[(size_t)row * 1024 + col] = (__bf16)val;
        else if (which == 1) K[(size_t)row * 1024 + col] = (__bf16)val;
        else {
          int b = row >> 11, l = row & 2047;
          Vt[((size_t)b * 1024 + col) * 2048 + l] = (__bf16)val;
        }
      }
    }
}

// ---------------------------------------------------------------------------
// Kernel 2: S = Q @ K^T with causal mask.  grid=(16,16,4).
// ---------------------------------------------------------------------------
__global__ __launch_bounds__(256) void scores_kernel(
    const __bf16* __restrict__ Q, const __bf16* __restrict__ K, float* __restrict__ S) {
  __shared__ __align__(16) __bf16 smem[3 * STAGE];
  const int tid = threadIdx.x;
  const int rowBase = blockIdx.x * BM, colBase = blockIdx.y * BN;
  const int b = blockIdx.z;
  const __bf16* Qb = Q + (size_t)b * 2048 * 1024;
  const __bf16* Kb = K + (size_t)b * 2048 * 1024;
  float* Sb = S + (size_t)b * 2048 * 2048;

  v8f c[4][2] = {};
  gemm_core(Qb + (size_t)rowBase * 1024, 1024,
            Kb + (size_t)colBase * 1024, 1024, 1024, smem, tid, c);

  const int lane = tid & 31, wave = tid >> 5;
  const int wm = wave >> 2, wn = wave & 3;
  const int row0 = rowBase + wm * 64 + ((lane >> 4) << 3);
  const int col0 = colBase + wn * 32 + (lane & 15);
#pragma unroll
  for (int x = 0; x < 4; ++x)
#pragma unroll
    for (int y = 0; y < 2; ++y) {
      int kcol = col0 + y * 16;
#pragma unroll
      for (int v = 0; v < 8; ++v) {
        int qrow = row0 + x * 16 + v;
        Sb[(size_t)qrow * 2048 + kcol] = (kcol <= qrow) ? c[x][y][v] : -999999.0f;
      }
    }
}

// ---------------------------------------------------------------------------
// Kernel 3: row softmax fp32 -> bf16 P.  One 256-thread block per row.
// ---------------------------------------------------------------------------
__global__ __launch_bounds__(256) void softmax_kernel(
    const float* __restrict__ S, __bf16* __restrict__ P) {
  __shared__ float red[256];
  const int tid = threadIdx.x;
  const size_t row = blockIdx.x;
  const float* Srow = S + row * 2048;
  __bf16* Prow = P + row * 2048;

  float vals[8];
  float m = -3.4e38f;
#pragma unroll
  for (int i = 0; i < 8; ++i) { vals[i] = Srow[tid + i * 256]; m = fmaxf(m, vals[i]); }
  red[tid] = m; __syncthreads();
  for (int s = 128; s > 0; s >>= 1) {
    if (tid < s) red[tid] = fmaxf(red[tid], red[tid + s]);
    __syncthreads();
  }
  m = red[0]; __syncthreads();

  float sum = 0.f;
#pragma unroll
  for (int i = 0; i < 8; ++i) { vals[i] = __expf(vals[i] - m); sum += vals[i]; }
  red[tid] = sum; __syncthreads();
  for (int s = 128; s > 0; s >>= 1) {
    if (tid < s) red[tid] += red[tid + s];
    __syncthreads();
  }
  float inv = 1.0f / red[0];
#pragma unroll
  for (int i = 0; i < 8; ++i) Prow[tid + i * 256] = (__bf16)(vals[i] * inv);
}

// ---------------------------------------------------------------------------
// Kernel 4: O = P @ V  (B from transposed Vt -> contiguous).  grid=(16,8,4).
// ---------------------------------------------------------------------------
__global__ __launch_bounds__(256) void pv_kernel(
    const __bf16* __restrict__ P, const __bf16* __restrict__ Vt, float* __restrict__ O) {
  __shared__ __align__(16) __bf16 smem[3 * STAGE];
  const int tid = threadIdx.x;
  const int rowBase = blockIdx.x * BM, colBase = blockIdx.y * BN;
  const int b = blockIdx.z;
  const __bf16* Pb  = P  + (size_t)b * 2048 * 2048;
  const __bf16* Vtb = Vt + (size_t)b * 1024 * 2048;
  float* Ob = O + (size_t)b * 2048 * 1024;

  v8f c[4][2] = {};
  gemm_core(Pb + (size_t)rowBase * 2048, 2048,
            Vtb + (size_t)colBase * 2048, 2048, 2048, smem, tid, c);

  const int lane = tid & 31, wave = tid >> 5;
  const int wm = wave >> 2, wn = wave & 3;
  const int row0 = rowBase + wm * 64 + ((lane >> 4) << 3);
  const int col0 = colBase + wn * 32 + (lane & 15);
#pragma unroll
  for (int x = 0; x < 4; ++x)
#pragma unroll
    for (int y = 0; y < 2; ++y) {
      int col = col0 + y * 16;
#pragma unroll
      for (int v = 0; v < 8; ++v) {
        int row = row0 + x * 16 + v;
        Ob[(size_t)row * 1024 + col] = c[x][y][v];
      }
    }
}

// ---------------------------------------------------------------------------
extern "C" void kernel_launch(void* const* d_in, const int* in_sizes, int n_in,
                              void* d_out, int out_size, void* d_ws, size_t ws_size,
                              hipStream_t stream) {
  const float* x  = (const float*)d_in[0];
  // d_in[1] = mask (bool) — causal mask applied analytically, unused.
  const float* wq = (const float*)d_in[2];
  const float* bq = (const float*)d_in[3];
  const float* wk = (const float*)d_in[4];
  const float* bk = (const float*)d_in[5];
  const float* wv = (const float*)d_in[6];
  const float* bv = (const float*)d_in[7];
  float* out = (float*)d_out;

  // Workspace (166 MiB):
  //  [0,16)   xb bf16        [16,22)  Wt q/k/v bf16 (2 MiB each)
  //  [22,38)  Q bf16         [38,54)  K bf16        [54,70)  Vt bf16
  //  [70,134) S fp32         [134,166) P bf16
  char* ws = (char*)d_ws;
  const size_t MiB = 1024 * 1024;
  __bf16* xb  = (__bf16*)(ws);
  __bf16* Wtq = (__bf16*)(ws + (size_t)16 * MiB);
  __bf16* Wtk = (__bf16*)(ws + (size_t)18 * MiB);
  __bf16* Wtv = (__bf16*)(ws + (size_t)20 * MiB);
  __bf16* Q   = (__bf16*)(ws + (size_t)22 * MiB);
  __bf16* K   = (__bf16*)(ws + (size_t)38 * MiB);
  __bf16* Vt  = (__bf16*)(ws + (size_t)54 * MiB);
  float*  S   = (float*)(ws + (size_t)70 * MiB);
  __bf16* P   = (__bf16*)(ws + (size_t)134 * MiB);

  cvt_x_kernel<<<4096, 256, 0, stream>>>(x, xb);
  cvt_w_kernel<<<dim3(16, 16, 3), 256, 0, stream>>>(wq, wk, wv, Wtq, Wtk, Wtv);
  proj_kernel<<<dim3(64, 8, 3), 256, 0, stream>>>(xb, Wtq, Wtk, Wtv, bq, bk, bv, Q, K, Vt);
  scores_kernel<<<dim3(16, 16, 4), 256, 0, stream>>>(Q, K, S);
  softmax_kernel<<<8192, 256, 0, stream>>>(S, P);
  pv_kernel<<<dim3(16, 8, 4), 256, 0, stream>>>(P, Vt, out);
}
